// EpisodicMemoryModule_19834158973374
// MI455X (gfx1250) — matrix operand order repaired and checked by reading, compiled, not run on hardware
//
#include <hip/hip_runtime.h>
#include <hip/hip_bf16.h>

#define Bv 128
#define Nv 256
#define Uv 1024
#define NBLK 32   // persistent grid: 32 blocks x 128 threads = 128 waves

typedef __attribute__((ext_vector_type(16))) __bf16 v16bf;
typedef __attribute__((ext_vector_type(8)))  float  v8f;

// ---------------- WMMA fragment loaders (layouts per CDNA5 ISA 7.12.2) -------

// A-matrix 16x32 bf16 from fp32 row-major source.
// lane r<16: row m0+r, elems 0..7 = K[k0..k0+7], elems 8..15 = K[k0+16..k0+23]
// lane r+16: same rows, K pattern shifted by +8.
__device__ __forceinline__ v16bf load_a_f32(const float* A, int lda, int m0, int k0) {
    int lane = threadIdx.x & 31;
    int half = lane >> 4;
    int r    = lane & 15;
    const float* p = A + (size_t)(m0 + r) * lda + k0 + half * 8;
    v16bf a;
#pragma unroll
    for (int i = 0; i < 8; ++i) {
        a[i]     = (__bf16)p[i];
        a[i + 8] = (__bf16)p[i + 16];
    }
    return a;
}

// Same, source already bf16 row-major.
__device__ __forceinline__ v16bf load_a_bf16(const __bf16* A, int lda, int m0, int k0) {
    int lane = threadIdx.x & 31;
    int half = lane >> 4;
    int r    = lane & 15;
    const __bf16* p = A + (size_t)(m0 + r) * lda + k0 + half * 8;
    v16bf a;
#pragma unroll
    for (int i = 0; i < 8; ++i) {
        a[i]     = p[i];
        a[i + 8] = p[i + 16];
    }
    return a;
}

// B-matrix 32x16 from pre-transposed bf16 weights Bt[n][k].
// lane c<16: column n0+c, elems 0..7 = K[k0..k0+7], elems 8..15 = K[k0+16..k0+23]; lane c+16: shifted by +8.
__device__ __forceinline__ v16bf load_b(const __bf16* Bt, int ldb, int n0, int k0) {
    int lane = threadIdx.x & 31;
    int half = lane >> 4;
    int c    = lane & 15;
    const __bf16* p = Bt + (size_t)(n0 + c) * ldb + k0 + half * 8;
    v16bf b;
#pragma unroll
    for (int i = 0; i < 8; ++i) {
        b[i]     = p[i];
        b[i + 8] = p[i + 16];
    }
    return b;
}

#define WMMA_BF16(a, b, acc) \
    __builtin_amdgcn_wmma_f32_16x16x32_bf16(false, (a), false, (b), (short)0, (acc), false, false)

// ---------------- device-scope grid barrier (generation counter in L2) ------------------
__device__ __forceinline__ void grid_sync(unsigned* cnt, unsigned* gen) {
    __threadfence();
    __syncthreads();
    if (threadIdx.x == 0) {
        unsigned g = __hip_atomic_load(gen, __ATOMIC_RELAXED, __HIP_MEMORY_SCOPE_AGENT);
        unsigned prev = __hip_atomic_fetch_add(cnt, 1u, __ATOMIC_ACQ_REL, __HIP_MEMORY_SCOPE_AGENT);
        if (prev == (unsigned)(gridDim.x - 1)) {
            __hip_atomic_store(cnt, 0u, __ATOMIC_RELAXED, __HIP_MEMORY_SCOPE_AGENT);
            __hip_atomic_fetch_add(gen, 1u, __ATOMIC_RELEASE, __HIP_MEMORY_SCOPE_AGENT);
        } else {
            while (__hip_atomic_load(gen, __ATOMIC_ACQUIRE, __HIP_MEMORY_SCOPE_AGENT) == g) {
                __builtin_amdgcn_s_sleep(2);
            }
        }
    }
    __syncthreads();
    __threadfence();
}

__global__ void barrier_init(unsigned* cnt, unsigned* gen) {
    if (threadIdx.x == 0) { *cnt = 0u; *gen = 0u; }
}

// ---------------- one-time weight prep ----------------
__global__ void conv_rk(const float* __restrict__ rk, __bf16* __restrict__ KRt,
                        __bf16* __restrict__ KHt) {
    int k = blockIdx.x * blockDim.x + threadIdx.x;
    int n = blockIdx.y;
    KRt[(size_t)n * Uv + k] = (__bf16)rk[(size_t)k * 2 * Uv + n];
    KHt[(size_t)n * Uv + k] = (__bf16)rk[(size_t)k * 2 * Uv + Uv + n];
}

__global__ void conv_mn(const float* __restrict__ mn, __bf16* __restrict__ Wt) {
    int k = blockIdx.x * blockDim.x + threadIdx.x;
    int n = blockIdx.y;
    Wt[(size_t)n * 3 * Uv + k] = (__bf16)mn[(size_t)k * Uv + n];
}

// ---------------- one-time: X{r,h} = facts @ K{r,h} + b{r,h} (bf16 out) -----------------
__global__ void facts_gemm(const float* __restrict__ A, const __bf16* __restrict__ Bt,
                           const float* __restrict__ bias, __bf16* __restrict__ Out) {
    int lane   = threadIdx.x & 31;
    int wid    = threadIdx.x >> 5;
    int nstrip = blockIdx.x * 4 + wid;
    int m0     = blockIdx.y * 16;
    int n0     = nstrip * 64;
    v8f acc[4] = {};
    for (int k0 = 0; k0 < Uv; k0 += 32) {
        v16bf a = load_a_f32(A, Uv, m0, k0);
#pragma unroll
        for (int t = 0; t < 4; ++t) {
            v16bf b = load_b(Bt, Uv, n0 + t * 16, k0);
            acc[t] = WMMA_BF16(a, b, acc[t]);
        }
    }
    int half = lane >> 4, c = lane & 15;
#pragma unroll
    for (int t = 0; t < 4; ++t)
#pragma unroll
        for (int v = 0; v < 8; ++v) {
            int row = m0 + v + half * 8;
            int col = n0 + t * 16 + c;
            Out[(size_t)row * Uv + col] = (__bf16)(acc[t][v] + bias[col]);
        }
}

// ---------------- persistent GRU + memory updates (all 3 steps, 1 launch) ---------------
__global__ void __launch_bounds__(128, 1)
gru_persistent(const float* __restrict__ question,
               const __bf16* __restrict__ KRt, const __bf16* __restrict__ KHt,
               const __bf16* __restrict__ Xr,  const __bf16* __restrict__ Xh,
               const __bf16* __restrict__ Wt,  const float* __restrict__ mbias,
               float* __restrict__ h, __bf16* __restrict__ u,
               float* __restrict__ mem0, float* __restrict__ mem1,
               float* __restrict__ out, unsigned* cnt, unsigned* gen) {
    const int lane = threadIdx.x & 31;
    const int wg   = blockIdx.x * 4 + (threadIdx.x >> 5);  // 0..127
    const int m0   = (wg & 7) * 16;                        // batch tile
    const int n0   = (wg >> 3) * 64;                       // unit strip
    const int half = lane >> 4, c = lane & 15;

    const float* mi = question;  // memory = q initially (no copy needed)
    float*       mo = mem0;

    for (int step = 0; step < 3; ++step) {
        const float* hread = mi;  // GRU carry init = memory
        for (int n = 0; n < Nv; ++n) {
            // phase 1: u = hard_sigmoid(Xr[:,n,:] + h@KR) * h   (att == 1 exactly)
            {
                v8f acc[4] = {};
                for (int k0 = 0; k0 < Uv; k0 += 32) {
                    v16bf a = load_a_f32(hread, Uv, m0, k0);
#pragma unroll
                    for (int t = 0; t < 4; ++t)
                        acc[t] = WMMA_BF16(a, load_b(KRt, Uv, n0 + t * 16, k0), acc[t]);
                }
                const __bf16* XrN = Xr + (size_t)n * Uv;
#pragma unroll
                for (int t = 0; t < 4; ++t)
#pragma unroll
                    for (int v = 0; v < 8; ++v) {
                        int bb = m0 + v + half * 8;
                        int j  = n0 + t * 16 + c;
                        float s = acc[t][v] + (float)XrN[(size_t)bb * (Nv * Uv) + j];
                        float r = fminf(fmaxf(0.2f * s + 0.5f, 0.0f), 1.0f);
                        u[(size_t)bb * Uv + j] = (__bf16)(r * hread[(size_t)bb * Uv + j]);
                    }
            }
            grid_sync(cnt, gen);
            // phase 2: h = sigmoid(Xh[:,n,:] + u@KH)
            {
                v8f acc[4] = {};
                for (int k0 = 0; k0 < Uv; k0 += 32) {
                    v16bf a = load_a_bf16(u, Uv, m0, k0);
#pragma unroll
                    for (int t = 0; t < 4; ++t)
                        acc[t] = WMMA_BF16(a, load_b(KHt, Uv, n0 + t * 16, k0), acc[t]);
                }
                const __bf16* XhN = Xh + (size_t)n * Uv;
#pragma unroll
                for (int t = 0; t < 4; ++t)
#pragma unroll
                    for (int v = 0; v < 8; ++v) {
                        int bb = m0 + v + half * 8;
                        int j  = n0 + t * 16 + c;
                        float s = acc[t][v] + (float)XhN[(size_t)bb * (Nv * Uv) + j];
                        h[(size_t)bb * Uv + j] = 1.0f / (1.0f + __expf(-s));
                    }
            }
            grid_sync(cnt, gen);
            hread = h;
        }
        // memory = relu([mem | episode | q] @ memory_net + memory_bias)
        {
            v8f acc[4] = {};
            for (int k0 = 0; k0 < 3 * Uv; k0 += 32) {
                const float* src;
                int kk;
                if (k0 < Uv)          { src = mi;       kk = k0; }
                else if (k0 < 2 * Uv) { src = h;        kk = k0 - Uv; }
                else                  { src = question; kk = k0 - 2 * Uv; }
                v16bf a = load_a_f32(src, Uv, m0, kk);
#pragma unroll
                for (int t = 0; t < 4; ++t)
                    acc[t] = WMMA_BF16(a, load_b(Wt, 3 * Uv, n0 + t * 16, k0), acc[t]);
            }
            float* dst = (step == 2) ? out : mo;  // last step writes d_out tiles directly
#pragma unroll
            for (int t = 0; t < 4; ++t)
#pragma unroll
                for (int v = 0; v < 8; ++v) {
                    int bb = m0 + v + half * 8;
                    int j  = n0 + t * 16 + c;
                    dst[(size_t)bb * Uv + j] = fmaxf(acc[t][v] + mbias[j], 0.0f);
                }
        }
        grid_sync(cnt, gen);
        mi = mo;
        mo = (mo == mem0) ? mem1 : mem0;  // identical swap sequence in every block
    }
}

// ---------------- host orchestration ----------------
extern "C" void kernel_launch(void* const* d_in, const int* in_sizes, int n_in,
                              void* d_out, int out_size, void* d_ws, size_t ws_size,
                              hipStream_t stream) {
    (void)in_sizes; (void)n_in; (void)out_size; (void)ws_size;
    const float* facts    = (const float*)d_in[0];  // (B,N,U)
    const float* question = (const float*)d_in[1];  // (B,U)
    // d_in[2..5]: l_1, bias_l1, l_2, bias_l2 — provably dead (softmax over size-1 axis => att==1)
    const float* rk    = (const float*)d_in[6];     // (U,2U)
    const float* bias  = (const float*)d_in[7];     // (2U,)
    const float* mnet  = (const float*)d_in[8];     // (3U,U)
    const float* mbias = (const float*)d_in[9];     // (U,)

    char* w = (char*)d_ws;
    auto alloc = [&](size_t bytes) {
        char* p = w;
        w += (bytes + 255) & ~(size_t)255;
        return p;
    };
    __bf16*   KRt  = (__bf16*)alloc((size_t)Uv * Uv * 2);       // 2 MB
    __bf16*   KHt  = (__bf16*)alloc((size_t)Uv * Uv * 2);       // 2 MB
    __bf16*   Wt   = (__bf16*)alloc((size_t)3 * Uv * Uv * 2);   // 6 MB
    __bf16*   Xr   = (__bf16*)alloc((size_t)Bv * Nv * Uv * 2);  // 64 MB
    __bf16*   Xh   = (__bf16*)alloc((size_t)Bv * Nv * Uv * 2);  // 64 MB
    float*    h    = (float*) alloc((size_t)Bv * Uv * 4);
    __bf16*   u    = (__bf16*)alloc((size_t)Bv * Uv * 2);
    float*    mem0 = (float*) alloc((size_t)Bv * Uv * 4);
    float*    mem1 = (float*) alloc((size_t)Bv * Uv * 4);
    unsigned* bar  = (unsigned*)alloc(256);                     // [0]=cnt, [1]=gen

    // one-time: weight transpose/convert + fact projections (fully parallel)
    conv_rk<<<dim3(Uv / 256, Uv), 256, 0, stream>>>(rk, KRt, KHt);
    conv_mn<<<dim3(3 * Uv / 256, Uv), 256, 0, stream>>>(mnet, Wt);
    facts_gemm<<<dim3(4, (Bv * Nv) / 16), 128, 0, stream>>>(facts, KRt, bias, Xr);
    facts_gemm<<<dim3(4, (Bv * Nv) / 16), 128, 0, stream>>>(facts, KHt, bias + Uv, Xh);
    barrier_init<<<1, 32, 0, stream>>>(bar, bar + 1);

    // one persistent launch for the entire serial recurrence (3 steps x 256 facts)
    gru_persistent<<<NBLK, 128, 0, stream>>>(question, KRt, KHt, Xr, Xh, Wt, mbias,
                                             h, u, mem0, mem1, (float*)d_out,
                                             bar, bar + 1);
}
